// LoopVisibleLSTM_57080115364719
// MI455X (gfx1250) — matrix operand order repaired and checked by reading, not compiled, not run on hardware
//
#include <hip/hip_runtime.h>

typedef _Float16 half_t;
typedef _Float16 v8h  __attribute__((ext_vector_type(8)));
typedef _Float16 v16h __attribute__((ext_vector_type(16)));
typedef float    v8f  __attribute__((ext_vector_type(8)));

#define T_SEQ  512
#define BATCH  32
#define DIM_D  512
#define DIM_H  1024
#define GATES  4096            // 4*H
#define MROWS  16384           // T*B
#define REC_BLOCKS 32

// ---- CDNA5 async global->LDS copies (ASYNCcnt), with safe fallback ---------
#if defined(__has_builtin)
# if __has_builtin(__builtin_amdgcn_global_load_async_to_lds_b128) && \
     __has_builtin(__builtin_amdgcn_s_wait_asynccnt)
#  define HAVE_ASYNC_LDS 1
# endif
#endif
#ifndef HAVE_ASYNC_LDS
# define HAVE_ASYNC_LDS 0
#endif

#if HAVE_ASYNC_LDS
typedef int v4i_t __attribute__((ext_vector_type(4)));
typedef __attribute__((address_space(1))) v4i_t as1_v4i;
typedef __attribute__((address_space(3))) v4i_t as3_v4i;
__device__ __forceinline__ void async_cp128(half_t* lds_dst, const half_t* gsrc) {
    __builtin_amdgcn_global_load_async_to_lds_b128(
        (as1_v4i*)gsrc, (as3_v4i*)lds_dst, 0, 0);
}
#endif

// ---------------------------------------------------------------------------
// Fragment loaders per CDNA5 WMMA 16x16x32 f16 VGPR layouts (ISA 7.12.2).
// A (16xK slice, row-major, ld in halves):
//   lanes 0-15 : row = lane,    K offsets {0..7, 16..23}
//   lanes 16-31: row = lane-16, K offsets {8..15, 24..31}
__device__ __forceinline__ v16h fragA_ld(const half_t* __restrict__ p, int ld) {
    const int lane = threadIdx.x & 31;
    const half_t* rp = p + (lane & 15) * ld + (lane >> 4) * 8;
    v8h lo = *(const v8h*)(rp);
    v8h hi = *(const v8h*)(rp + 16);
    return __builtin_shufflevector(lo, hi, 0,1,2,3,4,5,6,7,8,9,10,11,12,13,14,15);
}
// B (Kx16). Stored here as W row-major [N,K] so B[k][n] = W[n][k]:
//   lanes 0-15 : col = lane,    K offsets {0..7, 16..23}
//   lanes 16-31: col = lane-16, K offsets {8..15, 24..31}
__device__ __forceinline__ v16h fragB_ld(const half_t* __restrict__ p, int ld) {
    const int lane = threadIdx.x & 31;
    const half_t* rp = p + (lane & 15) * ld + (lane >> 4) * 8;
    v8h lo = *(const v8h*)(rp);
    v8h hi = *(const v8h*)(rp + 16);
    return __builtin_shufflevector(lo, hi, 0,1,2,3,4,5,6,7,8,9,10,11,12,13,14,15);
}

__device__ __forceinline__ v8f wmma_f16(v16h a, v16h b, v8f c) {
    return __builtin_amdgcn_wmma_f32_16x16x32_f16(false, a, false, b, (short)0, c, false, false);
}

__device__ __forceinline__ float fsigmoid(float x) { return 1.0f / (1.0f + __expf(-x)); }
__device__ __forceinline__ float ftanhf(float x)   { return 2.0f * fsigmoid(2.0f * x) - 1.0f; }

// ---------------------------------------------------------------------------
// fp32 -> f16 conversion, 8 elements/thread (all sizes are multiples of 2048)
__global__ void cvt_kernel(const float* __restrict__ src, half_t* __restrict__ dst, long n) {
    long i = ((long)blockIdx.x * blockDim.x + threadIdx.x) * 8;
    if (i + 8 <= n) {
        const float4* s = (const float4*)(src + i);
        float4 a = s[0], b = s[1];
        v8h o = { (half_t)a.x, (half_t)a.y, (half_t)a.z, (half_t)a.w,
                  (half_t)b.x, (half_t)b.y, (half_t)b.z, (half_t)b.w };
        *(v8h*)(dst + i) = o;
    }
}

// ---------------------------------------------------------------------------
// C_f16[M,N] = A_f16[M,K] @ W_f16[N,K]^T + bias1[N] (+ bias2[N])
// Block tile 128x64, 8 waves (256 thr): wave grid 4(M) x 2(N), 32x32 per wave.
// Double-buffered LDS, async global->LDS copies, one barrier per k-chunk.
// revT: remap A rows m=(t,b) -> (T-1-t, b) to fold the backward time reversal.
__global__ __launch_bounds__(256)
void wmma_gemm_f16(const half_t* __restrict__ A, const half_t* __restrict__ W,
                   const float* __restrict__ bias1, const float* __restrict__ bias2,
                   half_t* __restrict__ C, int M, int N, int K, int revT)
{
    __shared__ __align__(16) half_t smem[2 * 128 * 32 + 2 * 64 * 32];  // 24 KB
    // buffer bases computed arithmetically (no LDS-pointer const arrays)
    #define AS_BUF(b) (smem + (b) * (128 * 32))
    #define BS_BUF(b) (smem + 2 * 128 * 32 + (b) * (64 * 32))

    const int tid  = threadIdx.x;
    const int lane = tid & 31;
    const int wid  = tid >> 5;     // 0..7
    const int wm   = wid >> 1;     // 0..3
    const int wn   = wid & 1;      // 0..1
    const int bm   = blockIdx.x * 128;
    const int bn   = blockIdx.y * 64;

    // cooperative-load assignments (fixed per thread across the K loop)
    const int r0  = tid >> 2;      // 0..63
    const int seg = tid & 3;       // 0..3 (8-half segment)
    int grow0 = bm + r0;
    int grow1 = bm + 64 + r0;
    if (revT) {
        int t0 = grow0 >> 5, b0 = grow0 & 31;
        grow0 = ((T_SEQ - 1 - t0) << 5) | b0;
        int t1 = grow1 >> 5, b1 = grow1 & 31;
        grow1 = ((T_SEQ - 1 - t1) << 5) | b1;
    }
    const half_t* gA0 = A + (long)grow0 * K + seg * 8;
    const half_t* gA1 = A + (long)grow1 * K + seg * 8;
    const half_t* gB  = W + (long)(bn + r0) * K + seg * 8;

    auto issue = [&](int kc, int buf) {
        half_t* a0 = AS_BUF(buf) + r0 * 32 + seg * 8;
        half_t* a1 = AS_BUF(buf) + (64 + r0) * 32 + seg * 8;
        half_t* b  = BS_BUF(buf) + r0 * 32 + seg * 8;
#if HAVE_ASYNC_LDS
        async_cp128(a0, gA0 + kc * 32);
        async_cp128(a1, gA1 + kc * 32);
        async_cp128(b,  gB  + kc * 32);
#else
        *(v8h*)a0 = *(const v8h*)(gA0 + kc * 32);
        *(v8h*)a1 = *(const v8h*)(gA1 + kc * 32);
        *(v8h*)b  = *(const v8h*)(gB  + kc * 32);
#endif
    };

    v8f acc[2][2] = {};
    const int NK = K / 32;
    issue(0, 0);
    for (int kc = 0; kc < NK; ++kc) {
        const int cur = kc & 1;
#if HAVE_ASYNC_LDS
        __builtin_amdgcn_s_wait_asynccnt(0);   // my chunk-kc LDS writes done
#endif
        __syncthreads();                        // everyone's writes visible
        if (kc + 1 < NK) issue(kc + 1, cur ^ 1);

        const half_t* Ac = AS_BUF(cur);
        const half_t* Bc = BS_BUF(cur);
        v16h a0 = fragA_ld(Ac + (wm * 32 +  0) * 32, 32);
        v16h a1 = fragA_ld(Ac + (wm * 32 + 16) * 32, 32);
        v16h b0 = fragB_ld(Bc + (wn * 32 +  0) * 32, 32);
        v16h b1 = fragB_ld(Bc + (wn * 32 + 16) * 32, 32);
        acc[0][0] = wmma_f16(a0, b0, acc[0][0]);
        acc[0][1] = wmma_f16(a0, b1, acc[0][1]);
        acc[1][0] = wmma_f16(a1, b0, acc[1][0]);
        acc[1][1] = wmma_f16(a1, b1, acc[1][1]);
    }
    #undef AS_BUF
    #undef BS_BUF

    // epilogue: stage f16 tile in LDS, then coalesced global_store_b128
    __syncthreads();
    half_t* Ct = smem;                          // 128x64 halves = 16 KB
    const int cr = lane >> 4;
    const int cc = lane & 15;
    #pragma unroll
    for (int i = 0; i < 2; ++i) {
        #pragma unroll
        for (int j = 0; j < 2; ++j) {
            const int colt = wn * 32 + j * 16 + cc;
            float bsum = bias1 ? bias1[bn + colt] : 0.0f;
            if (bias2) bsum += bias2[bn + colt];
            #pragma unroll
            for (int r = 0; r < 8; ++r) {
                const int rowt = wm * 32 + i * 16 + r + 8 * cr;
                Ct[rowt * 64 + colt] = (half_t)(acc[i][j][r] + bsum);
            }
        }
    }
    __syncthreads();
    #pragma unroll
    for (int it = 0; it < 4; ++it) {
        const int v   = tid + 256 * it;         // 0..1023
        const int row = v >> 3;
        const int s8  = v & 7;
        *(v8h*)(C + (long)(bm + row) * N + bn + s8 * 8) =
            *(const v8h*)(Ct + row * 64 + s8 * 8);
    }
}

// ---------------------------------------------------------------------------
__global__ void init_state_kernel(const float* __restrict__ h0, const float* __restrict__ c0,
                                  half_t* __restrict__ hbuf, float* __restrict__ cbuf,
                                  unsigned* __restrict__ bar)
{
    int i = blockIdx.x * blockDim.x + threadIdx.x;
    if (i < BATCH * DIM_H) { hbuf[i] = (half_t)h0[i]; cbuf[i] = c0[i]; }
    if (i == 0) bar[0] = 0u;
}

// ---------------------------------------------------------------------------
// Persistent recurrence: 32 blocks x 128 thr = 128 waves, one (16x16)-within-H
// tile each (2 M-tiles x 64 N-tiles). Per step each wave runs K=1024 over all
// 4 gate blocks (128 WMMA), applies the LSTM cell in-register, then one grid
// barrier. h is ping-pong double-buffered (read t&1, write (t+1)&1).
__global__ __launch_bounds__(128)
void lstm_rec_kernel(const half_t* __restrict__ xg,   // [T,B,4H] f16
                     const half_t* __restrict__ Whh,  // [4H,H]   f16
                     half_t* __restrict__ hb0, half_t* __restrict__ hb1,
                     float* __restrict__ cbuf,        // [B,H]
                     half_t* __restrict__ outAct,     // [T,B,H] f16 or null
                     float*  __restrict__ outFin,     // [T,B,2H] f32 or null
                     int reverse, int dirOff, unsigned* __restrict__ bar)
{
    const int lane  = threadIdx.x & 31;
    const int wglob = blockIdx.x * 4 + (threadIdx.x >> 5);  // 0..127
    const int m0 = (wglob >> 6) * 16;    // batch-tile origin (0 or 16)
    const int n0 = (wglob & 63) * 16;    // h-column tile origin
    const int cr = lane >> 4;
    const int cc = lane & 15;

    const half_t* __restrict__ Wi = Whh + (long)(0 * DIM_H + n0) * DIM_H;
    const half_t* __restrict__ Wf = Whh + (long)(1 * DIM_H + n0) * DIM_H;
    const half_t* __restrict__ Wg = Whh + (long)(2 * DIM_H + n0) * DIM_H;
    const half_t* __restrict__ Wo = Whh + (long)(3 * DIM_H + n0) * DIM_H;

    for (int t = 0; t < T_SEQ; ++t) {
        const half_t* hin  = (t & 1) ? hb1 : hb0;
        half_t*       hout = (t & 1) ? hb0 : hb1;

        v8f ai = {}, af = {}, ag = {}, ao = {};
        for (int k0 = 0; k0 < DIM_H; k0 += 32) {
            v16h a  = fragA_ld(hin + (long)m0 * DIM_H + k0, DIM_H);
            v16h bi = fragB_ld(Wi + k0, DIM_H);
            v16h bf = fragB_ld(Wf + k0, DIM_H);
            v16h bg = fragB_ld(Wg + k0, DIM_H);
            v16h bo = fragB_ld(Wo + k0, DIM_H);
            ai = wmma_f16(a, bi, ai);
            af = wmma_f16(a, bf, af);
            ag = wmma_f16(a, bg, ag);
            ao = wmma_f16(a, bo, ao);
        }

        // prefetch next timestep's gate pre-activations (HBM stream) so they
        // are in flight across the cell math + grid barrier
        if (t + 1 < T_SEQ) {
            const long nx = (long)(t + 1) * BATCH * GATES
                          + (long)(m0 + 8 * cr) * GATES + n0 + cc;
            __builtin_prefetch(xg + nx,             0, 0);
            __builtin_prefetch(xg + nx +     DIM_H, 0, 0);
            __builtin_prefetch(xg + nx + 2 * DIM_H, 0, 0);
            __builtin_prefetch(xg + nx + 3 * DIM_H, 0, 0);
        }

        // fused LSTM cell on the accumulator fragments
        const long xbase = (long)t * BATCH * GATES;
        #pragma unroll
        for (int r = 0; r < 8; ++r) {
            const int row = m0 + r + 8 * cr;       // batch index
            const int col = n0 + cc;               // h index
            const long xi = xbase + (long)row * GATES + col;
            float gi = (float)xg[xi              ] + ai[r];
            float gf = (float)xg[xi +     DIM_H  ] + af[r];
            float gg = (float)xg[xi + 2 * DIM_H  ] + ag[r];
            float go = (float)xg[xi + 3 * DIM_H  ] + ao[r];
            const int ci = row * DIM_H + col;
            float cn = fsigmoid(gf) * cbuf[ci] + fsigmoid(gi) * ftanhf(gg);
            cbuf[ci] = cn;
            float h = fsigmoid(go) * ftanhf(cn);
            hout[ci] = (half_t)h;
            if (outAct) outAct[((long)t * BATCH + row) * DIM_H + col] = (half_t)h;
            if (outFin) {
                const int tw = reverse ? (T_SEQ - 1 - t) : t;
                outFin[((long)tw * BATCH + row) * (2 * DIM_H) + dirOff + col] = h;
            }
        }

        // grid-wide barrier: make this step's h visible before anyone reads it
        __syncthreads();
        if (threadIdx.x == 0) {
            __threadfence();
            __hip_atomic_fetch_add(bar, 1u, __ATOMIC_ACQ_REL, __HIP_MEMORY_SCOPE_AGENT);
            const unsigned need = (unsigned)REC_BLOCKS * (unsigned)(t + 1);
            while (__hip_atomic_load(bar, __ATOMIC_ACQUIRE, __HIP_MEMORY_SCOPE_AGENT) < need) {
                __builtin_amdgcn_s_sleep(1);
            }
        }
        __syncthreads();
    }
}

// ---------------------------------------------------------------------------
extern "C" void kernel_launch(void* const* d_in, const int* in_sizes, int n_in,
                              void* d_out, int out_size, void* d_ws, size_t ws_size,
                              hipStream_t stream)
{
    const float* x       = (const float*)d_in[0];
    const float* Winit_f = (const float*)d_in[1];
    const float* binit_f = (const float*)d_in[2];
    const float* Winit_b = (const float*)d_in[3];
    const float* binit_b = (const float*)d_in[4];
    const float* Wih_f   = (const float*)d_in[5];
    const float* Whh_f   = (const float*)d_in[6];
    const float* bih_f   = (const float*)d_in[7];
    const float* bhh_f   = (const float*)d_in[8];
    const float* Wih_b   = (const float*)d_in[9];
    const float* Whh_b   = (const float*)d_in[10];
    const float* bih_b   = (const float*)d_in[11];
    const float* bhh_b   = (const float*)d_in[12];
    const float* h0_f    = (const float*)d_in[13];
    const float* c0_f    = (const float*)d_in[14];
    const float* h0_b    = (const float*)d_in[15];
    const float* c0_b    = (const float*)d_in[16];

    char* ws = (char*)d_ws;
    size_t off = 0;
    auto alloc = [&](size_t bytes) -> void* {
        void* p = ws + off;
        off = (off + bytes + 255) & ~(size_t)255;
        return p;
    };

    half_t* x16    = (half_t*)alloc((size_t)MROWS * DIM_D * sizeof(half_t));
    half_t* Wi16f  = (half_t*)alloc((size_t)DIM_H * DIM_D * sizeof(half_t));
    half_t* Wi16b  = (half_t*)alloc((size_t)DIM_H * DIM_D * sizeof(half_t));
    half_t* Wihf16 = (half_t*)alloc((size_t)2 * GATES * DIM_H * sizeof(half_t));
    half_t* Whhf16 = (half_t*)alloc((size_t)2 * GATES * DIM_H * sizeof(half_t));
    half_t* Wihb16 = (half_t*)alloc((size_t)2 * GATES * DIM_H * sizeof(half_t));
    half_t* Whhb16 = (half_t*)alloc((size_t)2 * GATES * DIM_H * sizeof(half_t));
    half_t* actA   = (half_t*)alloc((size_t)MROWS * DIM_H * sizeof(half_t));
    half_t* actB   = (half_t*)alloc((size_t)MROWS * DIM_H * sizeof(half_t));
    half_t* xgbuf  = (half_t*)alloc((size_t)MROWS * GATES * sizeof(half_t));
    half_t* hb0    = (half_t*)alloc((size_t)BATCH * DIM_H * sizeof(half_t));
    half_t* hb1    = (half_t*)alloc((size_t)BATCH * DIM_H * sizeof(half_t));
    float*  cbuf   = (float*)alloc((size_t)BATCH * DIM_H * sizeof(float));
    unsigned* bar  = (unsigned*)alloc(256);

    auto cvt = [&](const float* s, half_t* d, long n) {
        cvt_kernel<<<dim3((unsigned)(n / 8 / 256)), dim3(256), 0, stream>>>(s, d, n);
    };
    cvt(x,       x16,    (long)MROWS * DIM_D);
    cvt(Winit_f, Wi16f,  (long)DIM_H * DIM_D);
    cvt(Winit_b, Wi16b,  (long)DIM_H * DIM_D);
    cvt(Wih_f,   Wihf16, (long)2 * GATES * DIM_H);
    cvt(Whh_f,   Whhf16, (long)2 * GATES * DIM_H);
    cvt(Wih_b,   Wihb16, (long)2 * GATES * DIM_H);
    cvt(Whh_b,   Whhb16, (long)2 * GATES * DIM_H);

    auto gemm = [&](const half_t* A, const half_t* W, const float* b1, const float* b2,
                    half_t* C, int M, int N, int K, int revT) {
        dim3 g((unsigned)(M / 128), (unsigned)(N / 64));
        wmma_gemm_f16<<<g, dim3(256), 0, stream>>>(A, W, b1, b2, C, M, N, K, revT);
    };

    for (int dir = 0; dir < 2; ++dir) {
        const half_t* Wi    = dir ? Wi16b  : Wi16f;
        const float*  bi    = dir ? binit_b : binit_f;
        const half_t* Wih16 = dir ? Wihb16 : Wihf16;
        const half_t* Whh16 = dir ? Whhb16 : Whhf16;
        const float*  bih   = dir ? bih_b  : bih_f;
        const float*  bhh   = dir ? bhh_b  : bhh_f;
        const float*  h0    = dir ? h0_b   : h0_f;
        const float*  c0    = dir ? c0_b   : c0_f;

        // init projection (time-reversed A rows for backward dir)
        gemm(x16, Wi, bi, nullptr, actA, MROWS, DIM_H, DIM_D, dir);

        for (int l = 0; l < 2; ++l) {
            const half_t* xs = (l == 0) ? actA : actB;
            gemm(xs, Wih16 + (size_t)l * GATES * DIM_H,
                 bih + (size_t)l * GATES, bhh + (size_t)l * GATES,
                 xgbuf, MROWS, GATES, DIM_H, 0);

            init_state_kernel<<<dim3(BATCH * DIM_H / 256), 256, 0, stream>>>(
                h0 + (size_t)l * BATCH * DIM_H, c0 + (size_t)l * BATCH * DIM_H,
                hb0, cbuf, bar);

            lstm_rec_kernel<<<dim3(REC_BLOCKS), dim3(128), 0, stream>>>(
                xgbuf, Whh16 + (size_t)l * GATES * DIM_H,
                hb0, hb1, cbuf,
                (l == 0) ? actB : (half_t*)nullptr,
                (l == 1) ? (float*)d_out : (float*)nullptr,
                dir, dir * DIM_H, bar);
        }
    }
}
